// VectorQuantizer_16939351016121
// MI455X (gfx1250) — matrix operand order repaired and checked by reading, not compiled, not run on hardware
//
#include <hip/hip_runtime.h>
#include <hip/hip_bf16.h>

typedef __attribute__((ext_vector_type(16))) _Float16 v16h;
typedef __attribute__((ext_vector_type(8)))  float    v8f;

#define NUM_EMB   8192
#define DIM       256
#define NROWS     (32 * 512)          // 16384 flattened input vectors
#define LDS_ROW   264                 // 256 halves + 8 pad (bank-conflict mitigation)
#define COLS_PER_CHUNK 64             // codebook rows staged in LDS per iteration

// ---------------------------------------------------------------------------
// f32 -> f16 conversion (grid-stride)
// ---------------------------------------------------------------------------
__global__ void vq_cvt_f16(const float* __restrict__ src, _Float16* __restrict__ dst, int n) {
    int i = blockIdx.x * blockDim.x + threadIdx.x;
    int stride = gridDim.x * blockDim.x;
    for (; i < n; i += stride) dst[i] = (_Float16)src[i];
}

// ---------------------------------------------------------------------------
// codebook: f32 -> f16 + per-row squared norm ||e||^2
// one block (256 threads) per codebook row
// ---------------------------------------------------------------------------
__global__ void vq_cb_prep(const float* __restrict__ cb, _Float16* __restrict__ cbh,
                           float* __restrict__ enorm) {
    const int row = blockIdx.x;
    const int t   = threadIdx.x;              // 0..255
    float v = cb[(size_t)row * DIM + t];
    cbh[(size_t)row * DIM + t] = (_Float16)v;
    float s = v * v;
    #pragma unroll
    for (int m = 16; m >= 1; m >>= 1) s += __shfl_xor(s, m, 32);
    __shared__ float ws[8];
    if ((t & 31) == 0) ws[t >> 5] = s;
    __syncthreads();
    if (t < 8) {
        float x = ws[t];
        #pragma unroll
        for (int m = 4; m >= 1; m >>= 1) x += __shfl_xor(x, m, 32);
        if (t == 0) enorm[row] = x;
    }
}

// ---------------------------------------------------------------------------
// Fused GEMM + argmin:
//   dist(row, col) = ||e_col||^2 - 2 * <x_row, e_col>   (||x||^2 dropped: row-constant)
// Workgroup: 256 threads = 8 waves; each wave owns 16 rows (A panel resident
// in VGPRs), workgroup covers 128 rows. Codebook streamed through LDS in
// 64-row chunks shared by all 8 waves. v_wmma_f32_16x16x32_f16, f32 accum.
// ---------------------------------------------------------------------------
__global__ __launch_bounds__(256)
void vq_argmin(const _Float16* __restrict__ X,     // [NROWS][DIM] f16
               const _Float16* __restrict__ CB,    // [NUM_EMB][DIM] f16
               const float*    __restrict__ enorm, // [NUM_EMB]
               int*            __restrict__ outIdx)// [NROWS]
{
    __shared__ _Float16 lds[COLS_PER_CHUNK * LDS_ROW]; // 33792 B

    const int t    = threadIdx.x;
    const int wave = t >> 5;
    const int lane = t & 31;
    const int half = lane >> 4;   // 0: lanes 0-15, 1: lanes 16-31
    const int l16  = lane & 15;

    const int rbase = blockIdx.x * 128 + wave * 16;

    // Resident A fragments: 8 fragments cover k = 0..255 for this wave's 16 rows.
    // 16-bit A layout (16x32): lane l16 = row M; half 0 holds K {f*32+0..7, f*32+16..23},
    // half 1 holds K {f*32+8..15, f*32+24..31} -> two 16B loads per fragment.
    union AF { v16h h; uint4 u[2]; };
    AF a[8];
    const _Float16* xrow = X + (size_t)(rbase + l16) * DIM;
    #pragma unroll
    for (int f = 0; f < 8; ++f) {
        a[f].u[0] = *(const uint4*)(xrow + f * 32 + half * 8);
        a[f].u[1] = *(const uint4*)(xrow + f * 32 + 16 + half * 8);
    }

    float minv[8];
    int   mini[8];
    #pragma unroll
    for (int j = 0; j < 8; ++j) { minv[j] = 3.4e38f; mini[j] = 0; }

    for (int cb0 = 0; cb0 < NUM_EMB; cb0 += COLS_PER_CHUNK) {
        // Cooperative stage: 64 codebook rows (64*256 f16 = 32 KB) -> LDS.
        // 2048 uint4 transfers / 256 threads = 8 each.
        {
            const uint4* gsrc = (const uint4*)(CB + (size_t)cb0 * DIM);
            #pragma unroll
            for (int e = 0; e < 8; ++e) {
                int idx = t * 8 + e;        // 0..2047
                int row = idx >> 5;         // 32 uint4 per codebook row
                int pos = idx & 31;
                uint4 v = gsrc[idx];
                *(uint4*)(&lds[row * LDS_ROW + pos * 8]) = v;
            }
        }
        __syncthreads();

        #pragma unroll
        for (int ct = 0; ct < 4; ++ct) {    // four 16-column WMMA tiles per chunk
            v8f c = {};
            const _Float16* brow = &lds[(ct * 16 + l16) * LDS_ROW];
            #pragma unroll
            for (int f = 0; f < 8; ++f) {
                // 16-bit B layout (32x16): lane l16 = column N; half 0 holds K {f*32+0..7, f*32+16..23},
                // half 1 holds K {f*32+8..15, f*32+24..31} (same order as A).
                AF b;
                b.u[0] = *(const uint4*)(brow + f * 32 + half * 8);
                b.u[1] = *(const uint4*)(brow + f * 32 + 16 + half * 8);
                c = __builtin_amdgcn_wmma_f32_16x16x32_f16(
                        false, a[f].h, false, b.h, (short)0, c, false, false);
            }
            // Epilogue: C layout -> lane holds column (cb0+ct*16+l16) for 8 rows
            // (VGPR j = row j + 8*half).
            const int col = cb0 + ct * 16 + l16;
            const float en = enorm[col];
            #pragma unroll
            for (int j = 0; j < 8; ++j) {
                float dist = en - 2.0f * c[j];
                if (dist < minv[j]) { minv[j] = dist; mini[j] = col; }
            }
        }
        __syncthreads();
    }

    // Min-reduce across the 16 lanes holding the same row (within each half);
    // ties -> smaller index (matches jnp.argmin first-min semantics).
    #pragma unroll
    for (int j = 0; j < 8; ++j) {
        #pragma unroll
        for (int m = 8; m >= 1; m >>= 1) {
            float ov = __shfl_xor(minv[j], m, 32);
            int   oi = __shfl_xor(mini[j], m, 32);
            if (ov < minv[j] || (ov == minv[j] && oi < mini[j])) {
                minv[j] = ov; mini[j] = oi;
            }
        }
    }
    if (l16 == 0) {
        #pragma unroll
        for (int j = 0; j < 8; ++j)
            outIdx[rbase + half * 8 + j] = mini[j];
    }
}

// ---------------------------------------------------------------------------
// Gather + straight-through output + per-row loss partial.
// out = x + (q - x) computed exactly as the reference does (fp32 rounding).
// ---------------------------------------------------------------------------
__global__ void vq_gather(const float* __restrict__ X, const float* __restrict__ CB,
                          const int* __restrict__ idx, float* __restrict__ out,
                          float* __restrict__ partial) {
    const int row = blockIdx.x;
    const int t   = threadIdx.x;      // 256 threads = DIM
    const int k   = idx[row];
    float q = CB[(size_t)k * DIM + t];
    float x = X[(size_t)row * DIM + t];
    out[(size_t)row * DIM + t] = x + (q - x);
    float d = q - x;
    float s = d * d;
    #pragma unroll
    for (int m = 16; m >= 1; m >>= 1) s += __shfl_xor(s, m, 32);
    __shared__ float ws[8];
    if ((t & 31) == 0) ws[t >> 5] = s;
    __syncthreads();
    if (t == 0) {
        float tot = 0.f;
        #pragma unroll
        for (int i = 0; i < 8; ++i) tot += ws[i];
        partial[row] = tot;
    }
}

// ---------------------------------------------------------------------------
// Deterministic final loss reduction: loss = 1.25 * mean((q-x)^2)
// (e_latent and q_latent are value-identical; stop_gradient is a no-op on values)
// ---------------------------------------------------------------------------
__global__ void vq_loss(const float* __restrict__ partial, float* __restrict__ loss) {
    const int t = threadIdx.x;        // 256 threads
    float s = 0.f;
    for (int i = t; i < NROWS; i += 256) s += partial[i];
    #pragma unroll
    for (int m = 16; m >= 1; m >>= 1) s += __shfl_xor(s, m, 32);
    __shared__ float ws[8];
    if ((t & 31) == 0) ws[t >> 5] = s;
    __syncthreads();
    if (t == 0) {
        float tot = 0.f;
        #pragma unroll
        for (int i = 0; i < 8; ++i) tot += ws[i];
        *loss = 1.25f * tot / (float)((size_t)NROWS * DIM);
    }
}

// ---------------------------------------------------------------------------
extern "C" void kernel_launch(void* const* d_in, const int* in_sizes, int n_in,
                              void* d_out, int out_size, void* d_ws, size_t ws_size,
                              hipStream_t stream) {
    const float* x  = (const float*)d_in[0];  // [32,512,256] f32
    const float* cb = (const float*)d_in[1];  // [8192,256]   f32

    char* ws = (char*)d_ws;
    size_t off = 0;
    _Float16* xh    = (_Float16*)(ws + off); off += (size_t)NROWS   * DIM * sizeof(_Float16); // 8 MB
    _Float16* cbh   = (_Float16*)(ws + off); off += (size_t)NUM_EMB * DIM * sizeof(_Float16); // 4 MB
    float*    enorm = (float*)   (ws + off); off += (size_t)NUM_EMB * sizeof(float);          // 32 KB
    int*      idx   = (int*)     (ws + off); off += (size_t)NROWS   * sizeof(int);            // 64 KB
    float*    part  = (float*)   (ws + off); off += (size_t)NROWS   * sizeof(float);          // 64 KB

    float* out  = (float*)d_out;
    float* loss = out + (size_t)NROWS * DIM;  // outputs concatenated: quantized_st, then loss

    vq_cvt_f16<<<1024, 256, 0, stream>>>(x, xh, NROWS * DIM);
    vq_cb_prep<<<NUM_EMB, 256, 0, stream>>>(cb, cbh, enorm);
    vq_argmin <<<NROWS / 128, 256, 0, stream>>>(xh, cbh, enorm, idx);
    vq_gather <<<NROWS, 256, 0, stream>>>(x, cb, idx, out, part);
    vq_loss   <<<1, 256, 0, stream>>>(part, loss);
}
